// FeatureIndividualStageMoE_41841571398177
// MI455X (gfx1250) — compile-verified, matrix operand order checked by reading, not run on hardware
//
#include <hip/hip_runtime.h>
#include <hip/hip_bf16.h>

#define N_TOK   16384
#define DM      128
#define DFE     16
#define DEH     160
#define DR      64
#define G_GRP   8
#define E_EXP   4
#define NF      16
#define NG      (N_TOK * G_GRP)

typedef __attribute__((ext_vector_type(16))) _Float16 v16h;
typedef __attribute__((ext_vector_type(8)))  _Float16 h8;
typedef __attribute__((ext_vector_type(8)))  float    v8f;

#define V8F_ZERO {0.f,0.f,0.f,0.f,0.f,0.f,0.f,0.f}

__device__ __forceinline__ float gelu_f(float x) {
  return 0.5f * x * (1.0f + erff(x * 0.70710678118654752440f));
}

__device__ __forceinline__ v8f wmma_f16(v16h a, v16h b, v8f c) {
  // D(16x16,f32) = A(16x32,f16) x B(32x16,f16) + C
  return __builtin_amdgcn_wmma_f32_16x16x32_f16(false, a, false, b, (short)0, c, false, false);
}

// Load a 16x32 f16 WMMA operand fragment from a row-major LDS buffer.
// Layout per CDNA5 ISA: lanes 0-15 hold row (lane&15) with K = k0+{0..7, 16..23},
// lanes 16-31 hold the same rows with K = k0+{8..15, 24..31}.
// Works for A (rows = M) and for B when the weight is stored transposed [N][K].
__device__ __forceinline__ v16h load_frag(const _Float16* buf, int ld, int row0, int k0) {
  int lane = threadIdx.x & 31;
  const _Float16* p = buf + (size_t)(row0 + (lane & 15)) * ld + k0 + ((lane >> 4) << 3);
  h8 lo = *(const h8*)(p);
  h8 hi = *(const h8*)(p + 16);
  v16h r;
#pragma unroll
  for (int i = 0; i < 8; ++i) { r[i] = lo[i]; r[i + 8] = hi[i]; }
  return r;
}

// ---- CDNA5 async global->LDS staging (ASYNCcnt path, no VGPR round-trip) ----
__device__ __forceinline__ void async_ld_b128(unsigned lds_off, unsigned long long gaddr) {
  asm volatile("global_load_async_to_lds_b128 %0, %1, off"
               :: "v"(lds_off), "v"(gaddr) : "memory");
}
__device__ __forceinline__ void wait_async() {
  asm volatile("s_wait_asynccnt 0x0" ::: "memory");
}
// count = number of f16 elements (multiple of 8); 16B per lane per instruction.
__device__ __forceinline__ void async_copy(_Float16* dst_lds, const _Float16* src, int count) {
  unsigned base = (unsigned)(uintptr_t)dst_lds; // flat-LDS aperture: addr[31:0] = LDS offset
  unsigned long long g = (unsigned long long)(uintptr_t)src;
  int nvec = count >> 3;
  for (int i = threadIdx.x; i < nvec; i += blockDim.x)
    async_ld_b128(base + (unsigned)i * 16u, g + (unsigned long long)i * 16ull);
}

// ---------------------------------------------------------------------------
// K1: fused pre_ln + he_ln; one wave per token
// ---------------------------------------------------------------------------
__global__ void __launch_bounds__(256) k_preln(
    const float* __restrict__ hidden,
    const float* __restrict__ pg, const float* __restrict__ pb,
    const float* __restrict__ hg, const float* __restrict__ hb,
    _Float16* __restrict__ hnorm16, _Float16* __restrict__ hcin16) {
  int wave = threadIdx.x >> 5;
  int lane = threadIdx.x & 31;
  int tok = blockIdx.x * 8 + wave;
  if (tok >= N_TOK) return;
  const float* x = hidden + (size_t)tok * DM;
  float v[4];
#pragma unroll
  for (int i = 0; i < 4; ++i) v[i] = x[lane + 32 * i];
  float s = v[0] + v[1] + v[2] + v[3];
#pragma unroll
  for (int m = 16; m >= 1; m >>= 1) s += __shfl_xor(s, m, 32);
  float mu = s * (1.0f / 128.0f);
  float q = 0.f;
#pragma unroll
  for (int i = 0; i < 4; ++i) { float d = v[i] - mu; q += d * d; }
#pragma unroll
  for (int m = 16; m >= 1; m >>= 1) q += __shfl_xor(q, m, 32);
  float rs = rsqrtf(q * (1.0f / 128.0f) + 1e-5f);
  float h[4];
#pragma unroll
  for (int i = 0; i < 4; ++i) {
    int c = lane + 32 * i;
    h[i] = (v[i] - mu) * rs * pg[c] + pb[c];
    hnorm16[(size_t)tok * DM + c] = (_Float16)h[i];
  }
  float s2 = h[0] + h[1] + h[2] + h[3];
#pragma unroll
  for (int m = 16; m >= 1; m >>= 1) s2 += __shfl_xor(s2, m, 32);
  float mu2 = s2 * (1.0f / 128.0f);
  float q2 = 0.f;
#pragma unroll
  for (int i = 0; i < 4; ++i) { float d = h[i] - mu2; q2 += d * d; }
#pragma unroll
  for (int m = 16; m >= 1; m >>= 1) q2 += __shfl_xor(q2, m, 32);
  float rs2 = rsqrtf(q2 * (1.0f / 128.0f) + 1e-5f);
#pragma unroll
  for (int i = 0; i < 4; ++i) {
    int c = lane + 32 * i;
    hcin16[(size_t)tok * DM + c] = (_Float16)((h[i] - mu2) * rs2 * hg[c] + hb[c]);
  }
}

// ---------------------------------------------------------------------------
// K2: weight convert f32 -> f16, transposed to [N][Kpad] rows (zero pad K>=R)
// ---------------------------------------------------------------------------
__global__ void k_trans(const float* __restrict__ src, _Float16* __restrict__ dst,
                        int R, int C, int Rp) {
  const float* s = src + (size_t)blockIdx.z * R * C;
  _Float16* d = dst + (size_t)blockIdx.z * C * Rp;
  int total = C * Rp;
  for (int i = blockIdx.x * blockDim.x + threadIdx.x; i < total; i += gridDim.x * blockDim.x) {
    int c = i / Rp, r = i - c * Rp;
    d[i] = (r < R) ? (_Float16)s[(size_t)r * C + c] : (_Float16)0.0f;
  }
}

// ---------------------------------------------------------------------------
// K3: hidden encoder: hc = gelu(hcin @ w1 + b1) @ w2 + b2   [N,128]->[N,64]
// ---------------------------------------------------------------------------
__global__ void __launch_bounds__(128) k_henc(
    const _Float16* __restrict__ hcin, const _Float16* __restrict__ w1T,
    const float* __restrict__ b1, const _Float16* __restrict__ w2T,
    const float* __restrict__ b2, float* __restrict__ hc) {
  __shared__ __align__(16) _Float16 sA[64 * 128];
  __shared__ __align__(16) _Float16 sB[64 * 128];
  __shared__ __align__(16) _Float16 sT[64 * 64];
  int base = blockIdx.x * 64;
  async_copy(sA, hcin + (size_t)base * 128, 64 * 128);
  async_copy(sB, w1T, 64 * 128);
  wait_async();
  __syncthreads();
  int wave = threadIdx.x >> 5, lane = threadIdx.x & 31;
  int mt = wave;
  int nloc = lane & 15;
  int mb = (lane >> 4) << 3;
#pragma unroll
  for (int nt = 0; nt < 4; ++nt) {
    v8f acc = V8F_ZERO;
#pragma unroll
    for (int k0 = 0; k0 < 128; k0 += 32)
      acc = wmma_f16(load_frag(sA, 128, mt * 16, k0), load_frag(sB, 128, nt * 16, k0), acc);
    int n = nt * 16 + nloc;
    float bias = b1[n];
#pragma unroll
    for (int r = 0; r < 8; ++r)
      sT[(size_t)(mt * 16 + mb + r) * 64 + n] = (_Float16)gelu_f(acc[r] + bias);
  }
  __syncthreads();
  async_copy(sB, w2T, 64 * 64);
  wait_async();
  __syncthreads();
#pragma unroll
  for (int nt = 0; nt < 4; ++nt) {
    v8f acc = V8F_ZERO;
#pragma unroll
    for (int k0 = 0; k0 < 64; k0 += 32)
      acc = wmma_f16(load_frag(sT, 64, mt * 16, k0), load_frag(sB, 64, nt * 16, k0), acc);
    int n = nt * 16 + nloc;
    float bias = b2[n];
#pragma unroll
    for (int r = 0; r < 8; ++r)
      hc[(size_t)(base + mt * 16 + mb + r) * 64 + n] = acc[r] + bias;
  }
}

// ---------------------------------------------------------------------------
// K4: scalar feature encoder + id-emb + LN (VALU, tiny); one thread per (n,g)
// outputs K-padded f16 rows [NG][32] (k 16..31 zeroed)
// ---------------------------------------------------------------------------
__global__ void __launch_bounds__(256) k_fenc(
    const float* __restrict__ feat, const int* __restrict__ sidx,
    const float* __restrict__ fs_w1, const float* __restrict__ fs_b1,
    const float* __restrict__ fs_w2, const float* __restrict__ fs_b2,
    const float* __restrict__ fid, const float* __restrict__ lg,
    const float* __restrict__ lb, _Float16* __restrict__ felnp) {
  int row = blockIdx.x * blockDim.x + threadIdx.x;
  if (row >= NG) return;
  int n = row >> 3, g = row & 7;
  float sf = feat[(size_t)n * NF + sidx[g]];
  float t[16];
#pragma unroll
  for (int j = 0; j < 16; ++j) t[j] = gelu_f(sf * fs_w1[j] + fs_b1[j]);
  float fe[16];
#pragma unroll
  for (int j2 = 0; j2 < 16; ++j2) {
    float a = fs_b2[j2] + fid[g * 16 + j2];
#pragma unroll
    for (int j = 0; j < 16; ++j) a += t[j] * fs_w2[j * 16 + j2];
    fe[j2] = a;
  }
  float mu = 0.f;
#pragma unroll
  for (int j = 0; j < 16; ++j) mu += fe[j];
  mu *= (1.0f / 16.0f);
  float var = 0.f;
#pragma unroll
  for (int j = 0; j < 16; ++j) { float d = fe[j] - mu; var += d * d; }
  float rs = rsqrtf(var * (1.0f / 16.0f) + 1e-5f);
  _Float16* o = felnp + (size_t)row * 32;
#pragma unroll
  for (int j = 0; j < 16; ++j) o[j] = (_Float16)((fe[j] - mu) * rs * lg[j] + lb[j]);
#pragma unroll
  for (int j = 16; j < 32; ++j) o[j] = (_Float16)0.0f;
}

// ---------------------------------------------------------------------------
// K5: feature router MLP: fc = gelu(feln @ fr_w1 + b1) @ fr_w2 + b2
// rows = NG, K = 16 (padded to 32) then 64
// ---------------------------------------------------------------------------
__global__ void __launch_bounds__(128) k_fr(
    const _Float16* __restrict__ felnp, const _Float16* __restrict__ w1T,
    const float* __restrict__ b1, const _Float16* __restrict__ w2T,
    const float* __restrict__ b2, float* __restrict__ fc) {
  __shared__ __align__(16) _Float16 sA[64 * 32];
  __shared__ __align__(16) _Float16 sB[64 * 64];
  __shared__ __align__(16) _Float16 sT[64 * 64];
  int base = blockIdx.x * 64;
  async_copy(sA, felnp + (size_t)base * 32, 64 * 32);
  async_copy(sB, w1T, 64 * 32);
  wait_async();
  __syncthreads();
  int wave = threadIdx.x >> 5, lane = threadIdx.x & 31;
  int mt = wave;
  int nloc = lane & 15;
  int mb = (lane >> 4) << 3;
#pragma unroll
  for (int nt = 0; nt < 4; ++nt) {
    v8f acc = V8F_ZERO;
    acc = wmma_f16(load_frag(sA, 32, mt * 16, 0), load_frag(sB, 32, nt * 16, 0), acc);
    int n = nt * 16 + nloc;
    float bias = b1[n];
#pragma unroll
    for (int r = 0; r < 8; ++r)
      sT[(size_t)(mt * 16 + mb + r) * 64 + n] = (_Float16)gelu_f(acc[r] + bias);
  }
  __syncthreads();
  async_copy(sB, w2T, 64 * 64);
  wait_async();
  __syncthreads();
#pragma unroll
  for (int nt = 0; nt < 4; ++nt) {
    v8f acc = V8F_ZERO;
#pragma unroll
    for (int k0 = 0; k0 < 64; k0 += 32)
      acc = wmma_f16(load_frag(sT, 64, mt * 16, k0), load_frag(sB, 64, nt * 16, k0), acc);
    int n = nt * 16 + nloc;
    float bias = b2[n];
#pragma unroll
    for (int r = 0; r < 8; ++r)
      fc[(size_t)(base + mt * 16 + mb + r) * 64 + n] = acc[r] + bias;
  }
}

// ---------------------------------------------------------------------------
// K6: compose + outer router + inner router logits.
// grid = (N/64, G); block = 128 (4 waves); 64KB LDS.
// ---------------------------------------------------------------------------
__global__ void __launch_bounds__(128) k_comp(
    const float* __restrict__ hc, const float* __restrict__ fc,
    const _Float16* __restrict__ orw1T, const float* __restrict__ orb1,
    const float* __restrict__ orw2, const float* __restrict__ orb2,
    const _Float16* __restrict__ irw1T, const float* __restrict__ irb1,
    const float* __restrict__ irw2, const float* __restrict__ irb2,
    float* __restrict__ flog, float* __restrict__ ilog) {
  __shared__ __align__(16) _Float16 sC[64 * 256];
  __shared__ __align__(16) _Float16 sB[64 * 256];
  int g = blockIdx.y;
  int base = blockIdx.x * 64;
  async_copy(sB, orw1T, 64 * 256);
  for (int i = threadIdx.x; i < 64 * 64; i += blockDim.x) {
    int r = i >> 6, j = i & 63;
    int tok = base + r;
    float f = fc[((size_t)tok * G_GRP + g) * 64 + j];
    float h = hc[(size_t)tok * 64 + j];
    _Float16* row = sC + (size_t)r * 256;
    row[j]       = (_Float16)f;
    row[64 + j]  = (_Float16)h;
    row[128 + j] = (_Float16)(f * h);
    row[192 + j] = (_Float16)fabsf(f - h);
  }
  wait_async();
  __syncthreads();
  int wave = threadIdx.x >> 5, lane = threadIdx.x & 31;
  int mt = wave;
  int nloc = lane & 15;
  int mb = (lane >> 4) << 3;
  // --- outer router: logits[m] = sum_n gelu(T[m,n]+b1[n]) * orw2[n] + orb2 ---
  float p[8];
#pragma unroll
  for (int r = 0; r < 8; ++r) p[r] = 0.f;
#pragma unroll
  for (int nt = 0; nt < 4; ++nt) {
    v8f acc = V8F_ZERO;
#pragma unroll
    for (int k0 = 0; k0 < 256; k0 += 32)
      acc = wmma_f16(load_frag(sC, 256, mt * 16, k0), load_frag(sB, 256, nt * 16, k0), acc);
    int n = nt * 16 + nloc;
    float bias = orb1[n], wv = orw2[n];
#pragma unroll
    for (int r = 0; r < 8; ++r) p[r] += gelu_f(acc[r] + bias) * wv;
  }
#pragma unroll
  for (int m = 1; m < 16; m <<= 1)
#pragma unroll
    for (int r = 0; r < 8; ++r) p[r] += __shfl_xor(p[r], m, 32);
  if ((lane & 15) == 0) {
#pragma unroll
    for (int r = 0; r < 8; ++r) {
      int mrow = mt * 16 + mb + r;
      flog[(size_t)(base + mrow) * G_GRP + g] = p[r] + orb2[0]; // TEMP = 1
    }
  }
  __syncthreads();
  async_copy(sB, irw1T + (size_t)g * 64 * 256, 64 * 256);
  wait_async();
  __syncthreads();
  // --- inner router: logits[m,e] = sum_n gelu(T[m,n]+b1) * irw2[n,e] + b2 ---
  float q[4][8];
#pragma unroll
  for (int e = 0; e < 4; ++e)
#pragma unroll
    for (int r = 0; r < 8; ++r) q[e][r] = 0.f;
#pragma unroll
  for (int nt = 0; nt < 4; ++nt) {
    v8f acc = V8F_ZERO;
#pragma unroll
    for (int k0 = 0; k0 < 256; k0 += 32)
      acc = wmma_f16(load_frag(sC, 256, mt * 16, k0), load_frag(sB, 256, nt * 16, k0), acc);
    int n = nt * 16 + nloc;
    float bias = irb1[g * 64 + n];
    float wv[4];
#pragma unroll
    for (int e = 0; e < 4; ++e) wv[e] = irw2[((size_t)g * 64 + n) * 4 + e];
#pragma unroll
    for (int r = 0; r < 8; ++r) {
      float il = gelu_f(acc[r] + bias);
#pragma unroll
      for (int e = 0; e < 4; ++e) q[e][r] += il * wv[e];
    }
  }
#pragma unroll
  for (int m = 1; m < 16; m <<= 1)
#pragma unroll
    for (int e = 0; e < 4; ++e)
#pragma unroll
      for (int r = 0; r < 8; ++r) q[e][r] += __shfl_xor(q[e][r], m, 32);
  if ((lane & 15) == 0) {
#pragma unroll
    for (int r = 0; r < 8; ++r) {
      int tok = base + mt * 16 + mb + r;
#pragma unroll
      for (int e = 0; e < 4; ++e)
        ilog[((size_t)tok * G_GRP + g) * E_EXP + e] = q[e][r] + irb2[g * 4 + e];
    }
  }
}

// ---------------------------------------------------------------------------
// K7: top-4 / top-2 softmax routing -> combined weights w[n, g*4+e]
// ---------------------------------------------------------------------------
__global__ void __launch_bounds__(256) k_route(
    const float* __restrict__ flog, const float* __restrict__ ilog,
    float* __restrict__ wcomb) {
  int n = blockIdx.x * blockDim.x + threadIdx.x;
  if (n >= N_TOK) return;
  float l[8];
#pragma unroll
  for (int g = 0; g < 8; ++g) l[g] = flog[(size_t)n * 8 + g];
  int used = 0;
  int idx[4]; float val[4];
#pragma unroll
  for (int k = 0; k < 4; ++k) {
    int bi = -1; float bv = -3.4e38f;
#pragma unroll
    for (int g = 0; g < 8; ++g)
      if (!((used >> g) & 1) && l[g] > bv) { bv = l[g]; bi = g; }
    used |= (1 << bi); idx[k] = bi; val[k] = bv;
  }
  float mx = val[0], se = 0.f, ev[4];
#pragma unroll
  for (int k = 0; k < 4; ++k) { ev[k] = expf(val[k] - mx); se += ev[k]; }
  float fw[8];
#pragma unroll
  for (int g = 0; g < 8; ++g) fw[g] = 0.f;
#pragma unroll
  for (int k = 0; k < 4; ++k) fw[idx[k]] = ev[k] / se;
  for (int g = 0; g < 8; ++g) {
    float il4[4];
#pragma unroll
    for (int e = 0; e < 4; ++e) il4[e] = ilog[((size_t)n * 8 + g) * 4 + e];
    int b0 = 0;
#pragma unroll
    for (int e = 1; e < 4; ++e) if (il4[e] > il4[b0]) b0 = e;
    int b1i = -1;
#pragma unroll
    for (int e = 0; e < 4; ++e)
      if (e != b0 && (b1i < 0 || il4[e] > il4[b1i])) b1i = e;
    float e1 = expf(il4[b1i] - il4[b0]);
    float s2 = 1.0f + e1;
    float iw[4];
#pragma unroll
    for (int e = 0; e < 4; ++e) iw[e] = 0.f;
    iw[b0] = 1.0f / s2; iw[b1i] = e1 / s2;
#pragma unroll
    for (int e = 0; e < 4; ++e)
      wcomb[((size_t)n * 8 + g) * 4 + e] = fw[g] * iw[e];
  }
}

// ---------------------------------------------------------------------------
// K8: experts. 32-token tile, 4 waves, loop over 32 (g,e) pairs.
// eh = gelu(h_norm @ ex_w1 + b1); eo = eh @ ex_w2 + b2;
// mixed += w[n,g,e] * eo; out = hidden + alpha * mixed.
// Weight staging via GLOBAL_LOAD_ASYNC_TO_LDS_B128 (ASYNCcnt).
// ---------------------------------------------------------------------------
__global__ void __launch_bounds__(128) k_expert(
    const _Float16* __restrict__ hnorm16, const float* __restrict__ hidden,
    const _Float16* __restrict__ ew1T, const float* __restrict__ eb1,
    const _Float16* __restrict__ ew2T, const float* __restrict__ eb2,
    const float* __restrict__ wcomb, const float* __restrict__ alpha_p,
    float* __restrict__ out) {
  __shared__ __align__(16) _Float16 sA[32 * 128];   // token activations (f16)
  __shared__ __align__(16) _Float16 sB[160 * 128];  // weight buffer (max of w1T/w2T)
  __shared__ __align__(16) _Float16 sE[32 * 160];   // gelu intermediate
  __shared__ float sW[32 * 32];                     // routing weights
  int base = blockIdx.x * 32;
  async_copy(sA, hnorm16 + (size_t)base * 128, 32 * 128);
  for (int i = threadIdx.x; i < 32 * 32; i += blockDim.x)
    sW[i] = wcomb[(size_t)base * 32 + i];
  int wave = threadIdx.x >> 5, lane = threadIdx.x & 31;
  int omt = wave >> 1;        // output m-tile (0..1)
  int onb = (wave & 1) * 4;   // output n-tile base (0 or 4)
  int nloc = lane & 15;
  int mb = (lane >> 4) << 3;
  v8f macc[4] = {V8F_ZERO, V8F_ZERO, V8F_ZERO, V8F_ZERO};
  __syncthreads();
  for (int ge = 0; ge < 32; ++ge) {
    async_copy(sB, ew1T + (size_t)ge * (DEH * DM), DEH * DM);
    if (ge + 1 < 32) {
      const _Float16* nxt = ew1T + (size_t)(ge + 1) * (DEH * DM);
      __builtin_prefetch(nxt + (size_t)threadIdx.x * 160, 0, 1);
    }
    wait_async();
    __syncthreads();
    // GEMM1: [32,128] x [128,160] -> gelu -> sE. 20 tiles, 5 per wave.
    for (int tt = 0; tt < 5; ++tt) {
      int t = wave * 5 + tt;
      int mt = t / 10, nt = t - mt * 10;
      v8f acc = V8F_ZERO;
#pragma unroll
      for (int k0 = 0; k0 < 128; k0 += 32)
        acc = wmma_f16(load_frag(sA, 128, mt * 16, k0), load_frag(sB, 128, nt * 16, k0), acc);
      int n = nt * 16 + nloc;
      float bias = eb1[(size_t)ge * DEH + n];
#pragma unroll
      for (int r = 0; r < 8; ++r)
        sE[(size_t)(mt * 16 + mb + r) * DEH + n] = (_Float16)gelu_f(acc[r] + bias);
    }
    __syncthreads();
    async_copy(sB, ew2T + (size_t)ge * (DM * DEH), DM * DEH);
    wait_async();
    __syncthreads();
    // GEMM2: [32,160] x [160,128]; each wave owns its 4 output tiles.
#pragma unroll
    for (int j = 0; j < 4; ++j) {
      int nt = onb + j;
      v8f acc = V8F_ZERO;
#pragma unroll
      for (int k0 = 0; k0 < 160; k0 += 32)
        acc = wmma_f16(load_frag(sE, DEH, omt * 16, k0), load_frag(sB, DEH, nt * 16, k0), acc);
      int n = nt * 16 + nloc;
      float bias = eb2[(size_t)ge * DM + n];
#pragma unroll
      for (int r = 0; r < 8; ++r) {
        float ws = sW[(omt * 16 + mb + r) * 32 + ge];
        macc[j][r] += ws * (acc[r] + bias);
      }
    }
    __syncthreads();
  }
  float alpha = alpha_p[0];
#pragma unroll
  for (int j = 0; j < 4; ++j) {
    int n = (onb + j) * 16 + nloc;
#pragma unroll
    for (int r = 0; r < 8; ++r) {
      int tok = base + omt * 16 + mb + r;
      out[(size_t)tok * DM + n] = hidden[(size_t)tok * DM + n] + alpha * macc[j][r];
    }
  }
}

// ---------------------------------------------------------------------------
extern "C" void kernel_launch(void* const* d_in, const int* in_sizes, int n_in,
                              void* d_out, int out_size, void* d_ws, size_t ws_size,
                              hipStream_t stream) {
  (void)in_sizes; (void)n_in; (void)out_size; (void)ws_size;
  const float* hidden = (const float*)d_in[0];
  const float* feat   = (const float*)d_in[1];
  const int*   sidx   = (const int*)d_in[2];
  const float* pre_g  = (const float*)d_in[3];
  const float* pre_b  = (const float*)d_in[4];
  const float* he_g   = (const float*)d_in[5];
  const float* he_b   = (const float*)d_in[6];
  const float* he_w1  = (const float*)d_in[7];
  const float* he_b1  = (const float*)d_in[8];
  const float* he_w2  = (const float*)d_in[9];
  const float* he_b2  = (const float*)d_in[10];
  const float* fs_w1  = (const float*)d_in[11];
  const float* fs_b1  = (const float*)d_in[12];
  const float* fs_w2  = (const float*)d_in[13];
  const float* fs_b2  = (const float*)d_in[14];
  const float* fid    = (const float*)d_in[15];
  const float* fr_g   = (const float*)d_in[16];
  const float* fr_b   = (const float*)d_in[17];
  const float* fr_w1  = (const float*)d_in[18];
  const float* fr_b1  = (const float*)d_in[19];
  const float* fr_w2  = (const float*)d_in[20];
  const float* fr_b2  = (const float*)d_in[21];
  const float* or_w1  = (const float*)d_in[22];
  const float* or_b1  = (const float*)d_in[23];
  const float* or_w2  = (const float*)d_in[24];
  const float* or_b2  = (const float*)d_in[25];
  const float* ir_w1  = (const float*)d_in[26];
  const float* ir_b1  = (const float*)d_in[27];
  const float* ir_w2  = (const float*)d_in[28];
  const float* ir_b2  = (const float*)d_in[29];
  const float* ex_w1  = (const float*)d_in[30];
  const float* ex_b1  = (const float*)d_in[31];
  const float* ex_w2  = (const float*)d_in[32];
  const float* ex_b2  = (const float*)d_in[33];
  const float* alpha  = (const float*)d_in[34];
  float* out = (float*)d_out;

  char* w = (char*)d_ws;
  size_t off = 0;
  auto alloc = [&](size_t bytes) -> char* {
    off = (off + 255) & ~(size_t)255;
    char* p = w + off; off += bytes; return p;
  };
  _Float16* hnorm16 = (_Float16*)alloc((size_t)N_TOK * DM * 2);
  _Float16* hcin16  = (_Float16*)alloc((size_t)N_TOK * DM * 2);
  float*    hc      = (float*)   alloc((size_t)N_TOK * DR * 4);
  _Float16* felnp   = (_Float16*)alloc((size_t)NG * 32 * 2);
  float*    fc      = (float*)   alloc((size_t)NG * DR * 4);
  float*    flog    = (float*)   alloc((size_t)N_TOK * G_GRP * 4);
  float*    ilog    = (float*)   alloc((size_t)N_TOK * G_GRP * E_EXP * 4);
  float*    wcomb   = (float*)   alloc((size_t)N_TOK * G_GRP * E_EXP * 4);
  _Float16* he_w1T  = (_Float16*)alloc((size_t)64 * 128 * 2);
  _Float16* he_w2T  = (_Float16*)alloc((size_t)64 * 64 * 2);
  _Float16* fr_w1T  = (_Float16*)alloc((size_t)64 * 32 * 2);
  _Float16* fr_w2T  = (_Float16*)alloc((size_t)64 * 64 * 2);
  _Float16* or_w1T  = (_Float16*)alloc((size_t)64 * 256 * 2);
  _Float16* ir_w1T  = (_Float16*)alloc((size_t)G_GRP * 64 * 256 * 2);
  _Float16* ex_w1T  = (_Float16*)alloc((size_t)32 * DEH * DM * 2);
  _Float16* ex_w2T  = (_Float16*)alloc((size_t)32 * DM * DEH * 2);

  auto gx = [](int total) { return (total + 255) / 256; };

  // 1) double-LN
  k_preln<<<N_TOK / 8, 256, 0, stream>>>(hidden, pre_g, pre_b, he_g, he_b,
                                         hnorm16, hcin16);
  // 2) weight transposes f32->f16 [N][Kpad]
  k_trans<<<dim3(gx(64 * 128), 1, 1), 256, 0, stream>>>(he_w1, he_w1T, 128, 64, 128);
  k_trans<<<dim3(gx(64 * 64), 1, 1), 256, 0, stream>>>(he_w2, he_w2T, 64, 64, 64);
  k_trans<<<dim3(gx(64 * 32), 1, 1), 256, 0, stream>>>(fr_w1, fr_w1T, 16, 64, 32);
  k_trans<<<dim3(gx(64 * 64), 1, 1), 256, 0, stream>>>(fr_w2, fr_w2T, 64, 64, 64);
  k_trans<<<dim3(gx(64 * 256), 1, 1), 256, 0, stream>>>(or_w1, or_w1T, 256, 64, 256);
  k_trans<<<dim3(gx(64 * 256), 1, G_GRP), 256, 0, stream>>>(ir_w1, ir_w1T, 256, 64, 256);
  k_trans<<<dim3(gx(DEH * DM), 1, 32), 256, 0, stream>>>(ex_w1, ex_w1T, DM, DEH, DM);
  k_trans<<<dim3(gx(DM * DEH), 1, 32), 256, 0, stream>>>(ex_w2, ex_w2T, DEH, DM, DEH);
  // 3) hidden encoder
  k_henc<<<N_TOK / 64, 128, 0, stream>>>(hcin16, he_w1T, he_b1, he_w2T, he_b2, hc);
  // 4) scalar feature encoder
  k_fenc<<<NG / 256, 256, 0, stream>>>(feat, sidx, fs_w1, fs_b1, fs_w2, fs_b2,
                                       fid, fr_g, fr_b, felnp);
  // 5) feature router MLP
  k_fr<<<NG / 64, 128, 0, stream>>>(felnp, fr_w1T, fr_b1, fr_w2T, fr_b2, fc);
  // 6) compose + router logits
  k_comp<<<dim3(N_TOK / 64, G_GRP), 128, 0, stream>>>(
      hc, fc, or_w1T, or_b1, or_w2, or_b2, ir_w1T, ir_b1, ir_w2, ir_b2, flog, ilog);
  // 7) routing softmax
  k_route<<<N_TOK / 256, 256, 0, stream>>>(flog, ilog, wcomb);
  // 8) experts + mix + residual
  k_expert<<<N_TOK / 32, 128, 0, stream>>>(hnorm16, hidden, ex_w1T, ex_b1,
                                           ex_w2T, ex_b2, wcomb, alpha, out);
}
